// MultiHeadLatentAttention_29738353557791
// MI455X (gfx1250) — compile-verified, matrix operand order checked by reading, not run on hardware
//
#include <hip/hip_runtime.h>
#include <stdint.h>

#define BB 2
#define SS 2048
#define DD 2048
#define HH 16
#define HD 128
#define LL 128

typedef __bf16 bf16;
typedef __attribute__((ext_vector_type(16))) __bf16 bf16x16;
typedef __attribute__((ext_vector_type(8)))  float  v8f;
typedef __attribute__((ext_vector_type(4)))  unsigned int v4u;
typedef __attribute__((ext_vector_type(4)))  int v4i;
typedef __attribute__((ext_vector_type(8)))  int v8i;

__device__ __forceinline__ v8f vzero() {
  v8f z;
#pragma unroll
  for (int i = 0; i < 8; ++i) z[i] = 0.f;
  return z;
}

__device__ __forceinline__ v8f wmma_bf(bf16x16 a, bf16x16 b, v8f c) {
  return __builtin_amdgcn_wmma_f32_16x16x32_bf16(false, a, false, b, (short)0, c,
                                                 false, false);
}

// 16x32 A (or Bt) fragment from row-major [row][K] bf16 source; kc selects 32-wide K chunk.
// Lane l: row = l&15, first 8 elems at K = kc*32 + (l>>4)*8, second 8 at +16.
__device__ __forceinline__ bf16x16 ldfrag(const bf16* base, int ld, int kc) {
  int lane = threadIdx.x & 31;
  const bf16* p = base + (lane & 15) * ld + kc * 32 + ((lane >> 4) << 3);
  bf16x16 o;
  *((float4*)&o)       = *((const float4*)p);
  *(((float4*)&o) + 1) = *((const float4*)(p + 16));
  return o;
}

// Same fragment but from fp32 source, converting to bf16 on the fly.
__device__ __forceinline__ bf16x16 ldfrag_f32(const float* base, int ld, int kc) {
  int lane = threadIdx.x & 31;
  const float* p = base + (lane & 15) * ld + kc * 32 + ((lane >> 4) << 3);
  float4 f0 = *((const float4*)p);
  float4 f1 = *((const float4*)(p + 4));
  float4 f2 = *((const float4*)(p + 16));
  float4 f3 = *((const float4*)(p + 20));
  bf16x16 o;
  o[0] = (bf16)f0.x; o[1] = (bf16)f0.y; o[2] = (bf16)f0.z; o[3] = (bf16)f0.w;
  o[4] = (bf16)f1.x; o[5] = (bf16)f1.y; o[6] = (bf16)f1.z; o[7] = (bf16)f1.w;
  o[8] = (bf16)f2.x; o[9] = (bf16)f2.y; o[10] = (bf16)f2.z; o[11] = (bf16)f2.w;
  o[12] = (bf16)f3.x; o[13] = (bf16)f3.y; o[14] = (bf16)f3.z; o[15] = (bf16)f3.w;
  return o;
}

__device__ __forceinline__ float redadd16(float v) {
  v += __shfl_xor(v, 1, 32); v += __shfl_xor(v, 2, 32);
  v += __shfl_xor(v, 4, 32); v += __shfl_xor(v, 8, 32);
  return v;
}
__device__ __forceinline__ float redmax16(float v) {
  v = fmaxf(v, __shfl_xor(v, 1, 32)); v = fmaxf(v, __shfl_xor(v, 2, 32));
  v = fmaxf(v, __shfl_xor(v, 4, 32)); v = fmaxf(v, __shfl_xor(v, 8, 32));
  return v;
}

// ---------------------------------------------------------------------------
// Tensor Data Mover: 2D tile (rows x cols bf16 elements) global -> LDS,
// optional +16B pad per 256B row (CDNA5 ISA).
// 6-arg builtin form: (g0 v4u, g1 v8i, g2 v4i, g3 v4i, v8i, i32 cpol).
// ---------------------------------------------------------------------------
#define HAVE_TDM __has_builtin(__builtin_amdgcn_tensor_load_to_lds)

__device__ __forceinline__ void tdm_load_2d(unsigned lds_off, const void* gptr,
                                            unsigned rows, unsigned cols,
                                            unsigned stride_elems, bool pad) {
#if HAVE_TDM
  unsigned long long ga = (unsigned long long)(uintptr_t)gptr;
  v4u g0;
  g0[0] = 1u;                                    // count=1, user mode
  g0[1] = lds_off;                               // lds byte address
  g0[2] = (unsigned)ga;                          // global addr lo
  g0[3] = (unsigned)((ga >> 32) & 0x1FFFFFFu) | (2u << 30);  // addr hi | type=2
  v8i g1;
  unsigned w0 = (1u << 16);                      // data_size = 2 bytes
  if (pad) w0 |= (1u << 20) | (5u << 22) | (3u << 25);  // pad 4 dwords per 64 dwords
  g1[0] = (int)w0;
  g1[1] = (int)(cols << 16);                     // tensor_dim0[15:0]
  g1[2] = (int)(rows << 16);                     // tensor_dim0 hi | tensor_dim1 lo
  g1[3] = (int)(cols << 16);                     // tensor_dim1 hi | tile_dim0
  g1[4] = (int)rows;                             // tile_dim1 | tile_dim2=0
  g1[5] = (int)stride_elems;                     // tensor_dim0_stride lo32
  g1[6] = 0;
  g1[7] = 0;
  v4i z4 = {0, 0, 0, 0};
  v8i z8;
#pragma unroll
  for (int i = 0; i < 8; ++i) z8[i] = 0;
  __builtin_amdgcn_tensor_load_to_lds(g0, g1, z4, z4, z8, 0);
#else
  (void)lds_off; (void)gptr; (void)rows; (void)cols; (void)stride_elems; (void)pad;
#endif
}

// ---------------------------------------------------------------------------
// Kernel 1: weight bf16 conversion + absorbedT = (Wq * Wuk^T)^T  per head
// ---------------------------------------------------------------------------
__global__ __launch_bounds__(256, 1) void k_prep(
    const float* __restrict__ Wq, const float* __restrict__ Wdkv,
    const float* __restrict__ Wuk, const float* __restrict__ Wuv,
    const float* __restrict__ Wkrope, const float* __restrict__ Wqrope,
    const float* __restrict__ Wo,
    bf16* __restrict__ wdkv_bf, bf16* __restrict__ wkr_bf,
    bf16* __restrict__ wqr_bf, bf16* __restrict__ wuvT_bf,
    bf16* __restrict__ wo_bf, bf16* __restrict__ absT_bf) {
  __shared__ bf16 s_wq[128 * 128];
  __shared__ bf16 s_wuk[128 * 128];
  int h = blockIdx.x;
  long base = (long)h * 16384;
  for (int i = threadIdx.x; i < 16384; i += 256) {
    wdkv_bf[base + i] = (bf16)Wdkv[base + i];
    wkr_bf[base + i]  = (bf16)Wkrope[base + i];
    wqr_bf[base + i]  = (bf16)Wqrope[base + i];
    wo_bf[base + i]   = (bf16)Wo[base + i];
    int d = i >> 7, l = i & 127;
    wuvT_bf[base + (long)d * 128 + l] = (bf16)Wuv[base + (long)l * 128 + d];
    s_wq[i]  = (bf16)Wq[base + i];
    s_wuk[i] = (bf16)Wuk[base + i];
  }
  __syncthreads();
  int w = threadIdx.x >> 5, lane = threadIdx.x & 31;
  int mbase = (lane >> 4) << 3, n15 = lane & 15;
#pragma unroll
  for (int j = 0; j < 8; ++j) {
    v8f c = vzero();
#pragma unroll
    for (int k = 0; k < 4; ++k)
      c = wmma_bf(ldfrag(s_wq + w * 16 * 128, 128, k),
                  ldfrag(s_wuk + j * 16 * 128, 128, k), c);
#pragma unroll
    for (int r = 0; r < 8; ++r)  // transpose on store: absT[l][d]
      absT_bf[base + (long)(j * 16 + n15) * 128 + w * 16 + mbase + r] = (bf16)c[r];
  }
}

// ---------------------------------------------------------------------------
// Kernel 2: ckv (GEMM + in-register LayerNorm) and k_r (GEMM + in-register RoPE)
// grid = (S/128, H*B), 8 waves of 16 rows each. No LDS.
// ---------------------------------------------------------------------------
__global__ __launch_bounds__(256, 1) void k_ckv_kr(
    const float* __restrict__ x, const bf16* __restrict__ wdkv,
    const bf16* __restrict__ wkr, const float* __restrict__ gamma,
    const float* __restrict__ beta, bf16* __restrict__ ckv_bf,
    bf16* __restrict__ kr_bf, float* __restrict__ ckv_out,
    float* __restrict__ kr_out) {
  int s0 = blockIdx.x * 128;
  int h = blockIdx.y >> 1, b = blockIdx.y & 1;
  int hb = h * BB + b;
  int w = threadIdx.x >> 5, lane = threadIdx.x & 31;
  int mbase = (lane >> 4) << 3, n15 = lane & 15;
  int srow0 = s0 + w * 16;
  const float* xa = x + ((long)b * SS + srow0) * DD + h * HD;

  bf16x16 a[4];
#pragma unroll
  for (int k = 0; k < 4; ++k) a[k] = ldfrag_f32(xa, DD, k);

  v8f c[8];
  const bf16* wd = wdkv + (long)h * LL * HD;
#pragma unroll
  for (int j = 0; j < 8; ++j) {
    c[j] = vzero();
#pragma unroll
    for (int k = 0; k < 4; ++k)
      c[j] = wmma_bf(a[k], ldfrag(wd + j * 16 * HD, HD, k), c[j]);
  }
  // LayerNorm over L (columns) per row, stats via cross-lane reduce.
  float mu[8], rstd[8];
#pragma unroll
  for (int r = 0; r < 8; ++r) {
    float p = 0.f, q = 0.f;
#pragma unroll
    for (int j = 0; j < 8; ++j) { float v = c[j][r]; p += v; q += v * v; }
    p = redadd16(p); q = redadd16(q);
    float m = p * (1.f / 128.f);
    mu[r] = m;
    rstd[r] = rsqrtf(q * (1.f / 128.f) - m * m + 1e-5f);
  }
  float g[8], be[8];
#pragma unroll
  for (int j = 0; j < 8; ++j) {
    g[j]  = gamma[h * LL + j * 16 + n15];
    be[j] = beta[h * LL + j * 16 + n15];
  }
  long obase = ((long)b * HH + h) * SS * LL;
  long wbase = (long)hb * SS * LL;
#pragma unroll
  for (int j = 0; j < 8; ++j) {
#pragma unroll
    for (int r = 0; r < 8; ++r) {
      float y = (c[j][r] - mu[r]) * rstd[r] * g[j] + be[j];
      long srow = srow0 + mbase + r;
      int col = j * 16 + n15;
      ckv_out[obase + srow * LL + col] = y;
      ckv_bf[wbase + srow * LL + col] = (bf16)y;
    }
  }
  // k_rope = rope(x * Wkrope^T)
  const bf16* wk = wkr + (long)h * LL * HD;
#pragma unroll
  for (int j = 0; j < 8; ++j) {
    c[j] = vzero();
#pragma unroll
    for (int k = 0; k < 4; ++k)
      c[j] = wmma_bf(a[k], ldfrag(wk + j * 16 * HD, HD, k), c[j]);
  }
#pragma unroll
  for (int jj = 0; jj < 4; ++jj) {  // cols jj*16+n and +64 live in tiles jj, jj+4
    int i = jj * 16 + n15;
    float inv = __expf(-(float)i * 0.14391157f);  // 10000^{-i/64}
#pragma unroll
    for (int r = 0; r < 8; ++r) {
      float ang = (float)(srow0 + mbase + r) * inv;
      float sn = __sinf(ang), cs = __cosf(ang);
      float x1 = c[jj][r], x2 = c[jj + 4][r];
      c[jj][r] = x1 * cs - x2 * sn;
      c[jj + 4][r] = x2 * cs + x1 * sn;
    }
  }
#pragma unroll
  for (int j = 0; j < 8; ++j) {
#pragma unroll
    for (int r = 0; r < 8; ++r) {
      long srow = srow0 + mbase + r;
      int col = j * 16 + n15;
      kr_out[obase + srow * LL + col] = c[j][r];
      kr_bf[wbase + srow * LL + col] = (bf16)c[j][r];
    }
  }
}

// ---------------------------------------------------------------------------
// Kernel 3: q1 = x*absorbedT^T ; q_r = rope(ckv*Wqrope^T + bq) ; vT = (ckv*Wuv)^T
// ---------------------------------------------------------------------------
__global__ __launch_bounds__(256, 1) void k_q_v(
    const float* __restrict__ x, const bf16* __restrict__ ckv_bf,
    const bf16* __restrict__ absT, const bf16* __restrict__ wqr,
    const bf16* __restrict__ wuvT, const float* __restrict__ bq,
    bf16* __restrict__ q1_bf, bf16* __restrict__ qr_bf,
    bf16* __restrict__ vT_bf) {
  int s0 = blockIdx.x * 128;
  int h = blockIdx.y >> 1, b = blockIdx.y & 1;
  int hb = h * BB + b;
  int w = threadIdx.x >> 5, lane = threadIdx.x & 31;
  int mbase = (lane >> 4) << 3, n15 = lane & 15;
  int srow0 = s0 + w * 16;

  const float* xa = x + ((long)b * SS + srow0) * DD + h * HD;
  const bf16* ca = ckv_bf + ((long)hb * SS + srow0) * LL;
  bf16x16 ax[4], ac[4];
#pragma unroll
  for (int k = 0; k < 4; ++k) { ax[k] = ldfrag_f32(xa, DD, k); ac[k] = ldfrag(ca, LL, k); }

  long wbase = (long)hb * SS * LL;
  // q1
  const bf16* at = absT + (long)h * LL * HD;
#pragma unroll
  for (int j = 0; j < 8; ++j) {
    v8f c = vzero();
#pragma unroll
    for (int k = 0; k < 4; ++k)
      c = wmma_bf(ax[k], ldfrag(at + j * 16 * HD, HD, k), c);
#pragma unroll
    for (int r = 0; r < 8; ++r)
      q1_bf[wbase + (long)(srow0 + mbase + r) * LL + j * 16 + n15] = (bf16)c[r];
  }
  // q_r
  v8f c[8];
  const bf16* wq = wqr + (long)h * HD * LL;
#pragma unroll
  for (int j = 0; j < 8; ++j) {
    c[j] = vzero();
#pragma unroll
    for (int k = 0; k < 4; ++k)
      c[j] = wmma_bf(ac[k], ldfrag(wq + j * 16 * LL, LL, k), c[j]);
    float bias = bq[h * HD + j * 16 + n15];
#pragma unroll
    for (int r = 0; r < 8; ++r) c[j][r] += bias;
  }
#pragma unroll
  for (int jj = 0; jj < 4; ++jj) {
    int i = jj * 16 + n15;
    float inv = __expf(-(float)i * 0.14391157f);
#pragma unroll
    for (int r = 0; r < 8; ++r) {
      float ang = (float)(srow0 + mbase + r) * inv;
      float sn = __sinf(ang), cs = __cosf(ang);
      float x1 = c[jj][r], x2 = c[jj + 4][r];
      c[jj][r] = x1 * cs - x2 * sn;
      c[jj + 4][r] = x2 * cs + x1 * sn;
    }
  }
#pragma unroll
  for (int j = 0; j < 8; ++j)
#pragma unroll
    for (int r = 0; r < 8; ++r)
      qr_bf[wbase + (long)(srow0 + mbase + r) * LL + j * 16 + n15] = (bf16)c[j][r];
  // vT (store transposed [d][t])
  const bf16* wv = wuvT + (long)h * HD * LL;
#pragma unroll
  for (int j = 0; j < 8; ++j) {
    v8f cv = vzero();
#pragma unroll
    for (int k = 0; k < 4; ++k)
      cv = wmma_bf(ac[k], ldfrag(wv + j * 16 * LL, LL, k), cv);
#pragma unroll
    for (int r = 0; r < 8; ++r)
      vT_bf[((long)hb * HD + j * 16 + n15) * SS + srow0 + mbase + r] = (bf16)cv[r];
  }
}

// ---------------------------------------------------------------------------
// Kernel 4: flash attention + output projection.
// LDS: [0,34816) ckv tile (128 x 272B via TDM pad), [34816,69632) kr tile,
//      [69632,104448) per-wave P/o staging (8 x 16 x 272B).
// ---------------------------------------------------------------------------
__global__ __launch_bounds__(256, 1) void k_attn(
    const bf16* __restrict__ q1_bf, const bf16* __restrict__ qr_bf,
    const bf16* __restrict__ ckv_bf, const bf16* __restrict__ kr_bf,
    const bf16* __restrict__ vT_bf, const bf16* __restrict__ wo_bf,
    float* __restrict__ out) {
  __shared__ __align__(16) unsigned char smem[104448];
  int s0 = blockIdx.x * 128;
  int h = blockIdx.y >> 1, b = blockIdx.y & 1;
  int hb = h * BB + b;
  int w = threadIdx.x >> 5, lane = threadIdx.x & 31;
  int mbase = (lane >> 4) << 3, n15 = lane & 15;
  int srow0 = s0 + w * 16;
  const bf16* q1a = q1_bf + ((long)hb * SS + srow0) * LL;
  const bf16* qra = qr_bf + ((long)hb * SS + srow0) * LL;
  const bf16* cl = (const bf16*)(smem);
  const bf16* kl = (const bf16*)(smem + 34816);
  bf16* pw = (bf16*)(smem + 69632 + w * 16 * 272);

  v8f o[8];
#pragma unroll
  for (int j = 0; j < 8; ++j) o[j] = vzero();
  float mrow[8], lrow[8];
#pragma unroll
  for (int r = 0; r < 8; ++r) { mrow[r] = -1e30f; lrow[r] = 0.f; }

  for (int t0 = 0; t0 <= s0; t0 += 128) {
    __syncthreads();
#if HAVE_TDM
    if (threadIdx.x < 32) {
      tdm_load_2d(0u, ckv_bf + ((long)hb * SS + t0) * LL, 128, 128, 128, true);
      tdm_load_2d(34816u, kr_bf + ((long)hb * SS + t0) * LL, 128, 128, 128, true);
      __builtin_amdgcn_s_wait_tensorcnt(0);
    }
#else
    for (int i = threadIdx.x; i < 2048; i += 256) {  // 128 rows x 16 chunks of 16B
      int row = i >> 4, ch = i & 15;
      *(float4*)(smem + row * 272 + ch * 16) =
          *(const float4*)(ckv_bf + ((long)hb * SS + t0 + row) * LL + ch * 8);
      *(float4*)(smem + 34816 + row * 272 + ch * 16) =
          *(const float4*)(kr_bf + ((long)hb * SS + t0 + row) * LL + ch * 8);
    }
#endif
    __syncthreads();

    v8f sc[8];
#pragma unroll
    for (int j = 0; j < 8; ++j) sc[j] = vzero();
#pragma unroll
    for (int k = 0; k < 4; ++k) {
      bf16x16 aq = ldfrag(q1a, LL, k);
#pragma unroll
      for (int j = 0; j < 8; ++j)
        sc[j] = wmma_bf(aq, ldfrag(cl + j * 16 * 136, 136, k), sc[j]);
      bf16x16 ar = ldfrag(qra, LL, k);
#pragma unroll
      for (int j = 0; j < 8; ++j)
        sc[j] = wmma_bf(ar, ldfrag(kl + j * 16 * 136, 136, k), sc[j]);
    }
    bool diag = (t0 == s0);
#pragma unroll
    for (int j = 0; j < 8; ++j)
#pragma unroll
      for (int r = 0; r < 8; ++r) {
        float v = sc[j][r] * 0.0625f;  // 1/sqrt(2L)
        if (diag && (t0 + j * 16 + n15) > (srow0 + mbase + r)) v = -1e30f;
        sc[j][r] = v;
      }
    float scale[8];
#pragma unroll
    for (int r = 0; r < 8; ++r) {
      float mx = sc[0][r];
#pragma unroll
      for (int j = 1; j < 8; ++j) mx = fmaxf(mx, sc[j][r]);
      mx = redmax16(mx);
      float mn = fmaxf(mrow[r], mx);
      scale[r] = __expf(mrow[r] - mn);
      mrow[r] = mn;
    }
#pragma unroll
    for (int j = 0; j < 8; ++j)
#pragma unroll
      for (int r = 0; r < 8; ++r) sc[j][r] = __expf(sc[j][r] - mrow[r]);
#pragma unroll
    for (int r = 0; r < 8; ++r) {
      float s = 0.f;
#pragma unroll
      for (int j = 0; j < 8; ++j) s += sc[j][r];
      s = redadd16(s);
      lrow[r] = lrow[r] * scale[r] + s;
    }
#pragma unroll
    for (int j = 0; j < 8; ++j)
#pragma unroll
      for (int r = 0; r < 8; ++r) o[j][r] *= scale[r];
    // stage P (C-layout -> row-major) for use as A fragments
#pragma unroll
    for (int j = 0; j < 8; ++j)
#pragma unroll
      for (int r = 0; r < 8; ++r)
        pw[(mbase + r) * 136 + j * 16 + n15] = (bf16)sc[j][r];
    asm volatile("s_wait_dscnt 0" ::: "memory");
    // o += P * V   (Bt = vT[d][t] from global, K-chunks over t)
    const bf16* vta = vT_bf + (long)hb * HD * SS + t0;
#pragma unroll
    for (int k = 0; k < 4; ++k) {
      bf16x16 ap = ldfrag(pw, 136, k);
#pragma unroll
      for (int j = 0; j < 8; ++j)
        o[j] = wmma_bf(ap, ldfrag(vta + (long)j * 16 * SS, SS, k), o[j]);
    }
  }
#pragma unroll
  for (int r = 0; r < 8; ++r) lrow[r] = 1.f / lrow[r];
#pragma unroll
  for (int j = 0; j < 8; ++j)
#pragma unroll
    for (int r = 0; r < 8; ++r) o[j][r] *= lrow[r];
  // stage o, then out = o * Wo^T
#pragma unroll
  for (int j = 0; j < 8; ++j)
#pragma unroll
    for (int r = 0; r < 8; ++r)
      pw[(mbase + r) * 136 + j * 16 + n15] = (bf16)o[j][r];
  asm volatile("s_wait_dscnt 0" ::: "memory");
  const bf16* wo = wo_bf + (long)h * HD * HD;
#pragma unroll
  for (int j = 0; j < 8; ++j) {
    v8f c = vzero();
#pragma unroll
    for (int k = 0; k < 4; ++k)
      c = wmma_bf(ldfrag(pw, 136, k), ldfrag(wo + j * 16 * HD, HD, k), c);
#pragma unroll
    for (int r = 0; r < 8; ++r)
      out[((long)b * SS + srow0 + mbase + r) * DD + h * HD + j * 16 + n15] = c[r];
  }
}

// ---------------------------------------------------------------------------
extern "C" void kernel_launch(void* const* d_in, const int* in_sizes, int n_in,
                              void* d_out, int out_size, void* d_ws, size_t ws_size,
                              hipStream_t stream) {
  const float* x      = (const float*)d_in[0];
  const float* Wq     = (const float*)d_in[1];
  const float* Wdkv   = (const float*)d_in[2];
  const float* Wuk    = (const float*)d_in[3];
  const float* Wuv    = (const float*)d_in[4];
  const float* Wkrope = (const float*)d_in[5];
  const float* Wqrope = (const float*)d_in[6];
  const float* bqrope = (const float*)d_in[7];
  const float* gamma  = (const float*)d_in[8];
  const float* beta   = (const float*)d_in[9];
  const float* Wo     = (const float*)d_in[10];

  float* out0 = (float*)d_out;                            // (B,S,D)
  float* out1 = out0 + (size_t)BB * SS * DD;              // ckv (B,H,S,L)
  float* out2 = out1 + (size_t)BB * HH * SS * LL;         // k_r (B,H,S,L)

  char* wsb = (char*)d_ws;
  size_t off = 0;
  auto alloc = [&](size_t bytes) { size_t o = off; off += (bytes + 255) & ~(size_t)255; return o; };
  const size_t WSMALL = (size_t)HH * LL * HD * sizeof(bf16);
  const size_t WBIG   = (size_t)HH * BB * SS * LL * sizeof(bf16);
  bf16* wdkv_bf = (bf16*)(wsb + alloc(WSMALL));
  bf16* wkr_bf  = (bf16*)(wsb + alloc(WSMALL));
  bf16* wqr_bf  = (bf16*)(wsb + alloc(WSMALL));
  bf16* wuvT_bf = (bf16*)(wsb + alloc(WSMALL));
  bf16* wo_bf   = (bf16*)(wsb + alloc(WSMALL));
  bf16* absT_bf = (bf16*)(wsb + alloc(WSMALL));
  bf16* ckv_bf  = (bf16*)(wsb + alloc(WBIG));
  bf16* kr_bf   = (bf16*)(wsb + alloc(WBIG));
  bf16* q1_bf   = (bf16*)(wsb + alloc(WBIG));
  bf16* qr_bf   = (bf16*)(wsb + alloc(WBIG));
  bf16* vT_bf   = (bf16*)(wsb + alloc(WBIG));

  k_prep<<<HH, 256, 0, stream>>>(Wq, Wdkv, Wuk, Wuv, Wkrope, Wqrope, Wo,
                                 wdkv_bf, wkr_bf, wqr_bf, wuvT_bf, wo_bf, absT_bf);
  dim3 grid(SS / 128, HH * BB);
  k_ckv_kr<<<grid, 256, 0, stream>>>(x, wdkv_bf, wkr_bf, gamma, beta,
                                     ckv_bf, kr_bf, out1, out2);
  k_q_v<<<grid, 256, 0, stream>>>(x, ckv_bf, absT_bf, wqr_bf, wuvT_bf, bqrope,
                                  q1_bf, qr_bf, vT_bf);
  k_attn<<<grid, 256, 0, stream>>>(q1_bf, qr_bf, ckv_bf, kr_bf, vT_bf, wo_bf, out0);
}
